// ClothGraphCNN2_74045236183235
// MI455X (gfx1250) — matrix operand order repaired and checked by reading, not compiled, not run on hardware
//
#include <hip/hip_runtime.h>

// ---------------------------------------------------------------------------
// ClothGraphCNN2 forward for MI455X (gfx1250, wave32).
//   * activations: bf16, channels-last (B, N, Cp) with Cp = ceil32(C), zero pad
//   * channel GEMMs: v_wmma_f32_16x16x32_bf16, 128x64 tile, 8 WMMA / K-step
//   * LDS staging: global_load_async_to_lds_b128 + s_wait_asynccnt (ASYNCcnt)
//   * adjacency: runtime-compressed <=16 nnz rows, coalesced channel gathers
//   * input layer factorized (broadcast structure), heads emit f32 ch-first
// ---------------------------------------------------------------------------

typedef __attribute__((ext_vector_type(16))) __bf16 v16bf;
typedef __attribute__((ext_vector_type(8)))  __bf16 v8bf;
typedef __attribute__((ext_vector_type(8)))  float  v8f;

#define BATCHN 16
#define N_BODY 1723
#define N_GARM 4000

// ---------------- weight pack: f32 -> bf16, K zero-padded to /32, opt transpose
__global__ __launch_bounds__(256)
void k_pack_w(const float* __restrict__ W, __bf16* __restrict__ Wp,
              int M, int K, int Kp, int transpose)
{
    int t = blockIdx.x * 256 + threadIdx.x;
    if (t >= M * Kp) return;
    int m = t / Kp, k = t - m * Kp;
    float v = 0.f;
    if (k < K) v = transpose ? W[(size_t)k * M + m] : W[(size_t)m * K + k];
    Wp[t] = (__bf16)v;
}

// ---------------- WMMA GEMM (channels-last)
// Y[(b*Nv+n)*CpY + m] = relu( W[m,:] . X[(b*Nv+n)*CpX + :] + bias[m] + res )
// Wp: M x Kp bf16 (zero padded). X pads are zero, so K loop is unguarded.
// 128 threads = 4 waves; tile 128(M) x 64(N); K-step 32; async LDS staging.
__global__ __launch_bounds__(128)
void k_gemm_wmma(const __bf16* __restrict__ Wp, const __bf16* __restrict__ X,
                 const float* __restrict__ bias, const __bf16* __restrict__ res,
                 __bf16* __restrict__ Y, float* __restrict__ Ycf,
                 int M, int Kp, int CpX, int CpY, int Nv, int relu)
{
    __shared__ __align__(16) __bf16 Xs[64][40];   // [vertex][k], 80B rows

    const int b    = blockIdx.z;
    const int n0   = blockIdx.x * 64;
    const int m0   = blockIdx.y * 128;
    const int tid  = threadIdx.x;
    const int wave = tid >> 5;
    const int lane = tid & 31;

    v8f acc[2][4] = {{v8f{}, v8f{}, v8f{}, v8f{}}, {v8f{}, v8f{}, v8f{}, v8f{}}};

    // A-fragment row pointers (two 64-row fragments per tile, 16 rows/wave)
    const __bf16* Wrow[2];
    #pragma unroll
    for (int f = 0; f < 2; ++f) {
        int mw = m0 + f * 64 + wave * 16 + (lane & 15);
        if (mw >= M) mw = M - 1;                  // clamp; stores zero pads
        Wrow[f] = Wp + (size_t)mw * Kp;
    }
    const int hiA   = (lane >> 4) * 8;            // A: halves 0..7 / 8..15
    const int nloc  = lane & 15;
    const int khalf = (lane >> 4) * 8;            // B: halves 0..7 / 8..15

    for (int kk = 0; kk < Kp; kk += 32) {
        // ---- async stage: 64 rows x 32 halves (64B/row), 2 x b128 per thread
        #pragma unroll
        for (int q = 0; q < 2; ++q) {
            int chunk = tid + q * 128;            // 0..255
            int row   = chunk >> 2;               // 0..63
            int cpart = (chunk & 3) * 8;          // halves 0/8/16/24
            int n = n0 + row;
            if (n >= Nv) n = Nv - 1;              // duplicate; col-guarded later
            const __bf16* gsrc = X + ((size_t)b * Nv + n) * CpX + kk + cpart;
            unsigned lds = (unsigned)(unsigned long long)&Xs[row][cpart];
            unsigned long long ga = (unsigned long long)gsrc;
            asm volatile("global_load_async_to_lds_b128 %0, %1, off"
                         :: "v"(lds), "v"(ga) : "memory");
        }
        asm volatile("s_wait_asynccnt 0x0" ::: "memory");
        __syncthreads();

        // ---- A fragments (two contiguous 16B chunks each, L2-resident)
        v16bf a[2];
        #pragma unroll
        for (int f = 0; f < 2; ++f) {
            const __bf16* wp = Wrow[f] + kk;
            *(v8bf*)&a[f]       = *(const v8bf*)(wp + hiA);
            *((v8bf*)&a[f] + 1) = *(const v8bf*)(wp + 16 + hiA);
        }
        // ---- B fragments + 8 WMMAs
        #pragma unroll
        for (int c = 0; c < 4; ++c) {
            v16bf bb;
            const __bf16* xp = &Xs[c * 16 + nloc][khalf];
            *(v8bf*)&bb       = *(const v8bf*)xp;
            *((v8bf*)&bb + 1) = *(const v8bf*)(xp + 16);
            acc[0][c] = __builtin_amdgcn_wmma_f32_16x16x32_bf16(
                false, a[0], false, bb, (short)0, acc[0][c], false, false);
            acc[1][c] = __builtin_amdgcn_wmma_f32_16x16x32_bf16(
                false, a[1], false, bb, (short)0, acc[1][c], false, false);
        }
        __syncthreads();
    }

    // ---- epilogue: packed bf16 channels-last store (+ optional f32 ch-first)
    #pragma unroll
    for (int f = 0; f < 2; ++f) {
        #pragma unroll
        for (int c = 0; c < 4; ++c) {
            int n = n0 + c * 16 + nloc;
            if (n >= Nv) continue;
            int mb = m0 + f * 64 + wave * 16 + (lane >> 4) * 8;
            if (mb >= CpY) continue;
            size_t rowoff = ((size_t)b * Nv + n) * CpY + mb;
            float vals[8];
            #pragma unroll
            for (int r = 0; r < 8; ++r) {
                int m = mb + r;
                float v = 0.f;
                if (m < M) {
                    v = acc[f][c][r];
                    if (bias) v += bias[m];
                    if (relu && v < 0.f) v = 0.f;  // relu before residual? no:
                }
                vals[r] = v;
            }
            if (res) {
                v8bf rv = *(const v8bf*)(res + rowoff);
                #pragma unroll
                for (int r = 0; r < 8; ++r)
                    if (mb + r < M) vals[r] += (float)rv[r];
            }
            if (relu) {   // (applied once more is idempotent w/o residual)
                #pragma unroll
                for (int r = 0; r < 8; ++r) if (vals[r] < 0.f) vals[r] = 0.f;
            }
            if (Y) {
                __bf16 tmp[8];
                #pragma unroll
                for (int r = 0; r < 8; ++r) tmp[r] = (__bf16)vals[r];
                *(v8bf*)(Y + rowoff) = *(v8bf*)tmp;
            }
            if (Ycf) {
                #pragma unroll
                for (int r = 0; r < 8; ++r)
                    if (mb + r < M)
                        Ycf[((size_t)b * M + mb + r) * Nv + n] = vals[r];
            }
        }
    }
}

// ---------------- group norm over (8 ch x N) per (b,group); bf16 channels-last
__global__ __launch_bounds__(256)
void k_groupnorm(const __bf16* __restrict__ x, __bf16* __restrict__ y,
                 const float* __restrict__ gamma, const float* __restrict__ beta,
                 int C, int Cp, int N, int groups, int relu)
{
    int g = blockIdx.x, b = blockIdx.y;
    int c0 = g * 8;
    int cnt = 8 * N;
    float s = 0.f, ss = 0.f;
    for (int i = threadIdx.x; i < cnt; i += 256) {
        int n = i >> 3, c = c0 + (i & 7);
        float v = (float)x[((size_t)b * N + n) * Cp + c];
        s += v; ss += v * v;
    }
    __shared__ float rs[256], rq[256];
    rs[threadIdx.x] = s; rq[threadIdx.x] = ss;
    __syncthreads();
    for (int st = 128; st > 0; st >>= 1) {
        if (threadIdx.x < st) { rs[threadIdx.x] += rs[threadIdx.x + st];
                                rq[threadIdx.x] += rq[threadIdx.x + st]; }
        __syncthreads();
    }
    float mean = rs[0] / cnt;
    float var  = rq[0] / cnt - mean * mean;
    float rstd = rsqrtf(var + 1e-5f);
    for (int i = threadIdx.x; i < cnt; i += 256) {
        int n = i >> 3, c = c0 + (i & 7);
        size_t o = ((size_t)b * N + n) * Cp + c;
        float v = ((float)x[o] - mean) * rstd * gamma[c] + beta[c];
        if (relu && v < 0.f) v = 0.f;
        y[o] = (__bf16)v;
    }
    // last group also zeroes the channel padding (GEMM staging reads it)
    if (g == groups - 1 && Cp > C) {
        int pw = Cp - C;
        for (int i = threadIdx.x; i < N * pw; i += 256) {
            int n = i / pw, c = C + i % pw;
            y[((size_t)b * N + n) * Cp + c] = (__bf16)0.f;
        }
    }
}

// ---------------- adjacency compress: dense row -> <=16 (idx,val) pairs
__global__ __launch_bounds__(128)
void k_adj_compress(const float* __restrict__ A, int N,
                    int* __restrict__ idx, float* __restrict__ val)
{
    int m = blockIdx.x * 128 + threadIdx.x;
    if (m >= N) return;
    int cnt = 0;
    for (int n = 0; n < N; ++n) {
        float a = A[(size_t)m * N + n];
        if (a != 0.f && cnt < 16) { idx[m * 16 + cnt] = n; val[m * 16 + cnt] = a; ++cnt; }
    }
    for (; cnt < 16; ++cnt) { idx[m * 16 + cnt] = 0; val[m * 16 + cnt] = 0.f; }
}

// ---------------- sparse message passing, channels-last (coalesced over o)
__global__ __launch_bounds__(128)
void k_spmm(const __bf16* __restrict__ sup, const int* __restrict__ idx,
            const float* __restrict__ val, const float* __restrict__ bias,
            __bf16* __restrict__ out, int O, int Cp, int N)
{
    int o = blockIdx.x * 128 + threadIdx.x;
    if (o >= O) return;
    int m = blockIdx.y, b = blockIdx.z;
    float acc = bias[o];
    #pragma unroll
    for (int j = 0; j < 16; ++j) {
        int   ix = idx[m * 16 + j];
        float vv = val[m * 16 + j];
        acc = fmaf(vv, (float)sup[((size_t)b * N + ix) * Cp + o], acc);
    }
    out[((size_t)b * N + m) * Cp + o] = (__bf16)acc;
}

// ---------------- input-layer factorization
// R[n,o] = b0[o] + W0[o,0:3] . refv[:,n]        (f32, vertex-major)
__global__ __launch_bounds__(256)
void k_refproj(const float* __restrict__ W0, const float* __restrict__ b0,
               const float* __restrict__ refv, float* __restrict__ R, int N)
{
    int t = blockIdx.x * 256 + threadIdx.x;
    if (t >= N * 1024) return;
    int o = t & 1023, n = t >> 10;
    const float* w = W0 + (size_t)o * 2051;
    R[t] = b0[o] + w[0] * refv[n] + w[1] * refv[N + n] + w[2] * refv[2 * N + n];
}

// I[b,o] = W0[o,3:2051] . img[b,:]
__global__ __launch_bounds__(128)
void k_imgproj(const float* __restrict__ W0, const float* __restrict__ img,
               float* __restrict__ I)
{
    int o = blockIdx.x * 128 + threadIdx.x;
    int b = blockIdx.y;
    if (o >= 1024) return;
    const float* w  = W0 + (size_t)o * 2051 + 3;
    const float* im = img + (size_t)b * 2048;
    float s = 0.f;
    for (int c = 0; c < 2048; ++c) s = fmaf(w[c], im[c], s);
    I[b * 1024 + o] = s;
}

// X0[(b*N+n)*1024 + o] = bf16( R[n,o] + I[b,o] )
__global__ __launch_bounds__(256)
void k_bcast_add(const float* __restrict__ R, const float* __restrict__ I,
                 __bf16* __restrict__ X, int N, size_t total)
{
    size_t t = (size_t)blockIdx.x * 256 + threadIdx.x;
    if (t >= total) return;
    size_t o  = t & 1023;
    size_t bn = t >> 10;
    size_t n  = bn % (size_t)N;
    size_t b  = bn / (size_t)N;
    X[t] = (__bf16)(R[n * 1024 + o] + I[b * 1024 + o]);
}

// ---------------- camera fc: out[b,i] = bias[i] + sum_n W[i,n] * c[b,n]
__global__ __launch_bounds__(256)
void k_reduce_fc(const __bf16* __restrict__ c, const float* __restrict__ W,
                 const float* __restrict__ bias, float* __restrict__ out, int N)
{
    int i = blockIdx.x, b = blockIdx.y;
    __shared__ float red[256];
    const float* w = W + (size_t)i * N;
    float s = 0.f;
    for (int n = threadIdx.x; n < N; n += 256)
        s = fmaf(w[n], (float)c[((size_t)b * N + n) * 32], s);  // CC stride 32
    red[threadIdx.x] = s;
    __syncthreads();
    for (int st = 128; st > 0; st >>= 1) {
        if (threadIdx.x < st) red[threadIdx.x] += red[threadIdx.x + st];
        __syncthreads();
    }
    if (threadIdx.x == 0) out[b * 3 + i] = red[0] + bias[i];
}

// ---------------------------------------------------------------------------
struct GLp { const float *W, *b; };
struct GNp { const float *g, *be; };
struct RBp { GNp pre; GLp lin1; GNp n1; GLp conv; GNp n2; GLp lin2; GLp skip; int has_skip; };

extern "C" void kernel_launch(void* const* d_in, const int* in_sizes, int n_in,
                              void* d_out, int out_size, void* d_ws, size_t ws_size,
                              hipStream_t stream)
{
    (void)in_sizes; (void)n_in; (void)out_size; (void)ws_size;

    // ---- unpack inputs in setup_inputs() insertion order (depth-first leaves)
    int ii = 0;
    auto nx = [&]() { return (const float*)d_in[ii++]; };
    const float* img = nx();
    auto getGL = [&]() { GLp g; g.W = nx(); g.b = nx(); return g; };
    auto getGN = [&]() { GNp g; g.g = nx(); g.be = nx(); return g; };
    auto getRB = [&](int skip) {
        RBp r; r.pre = getGN(); r.lin1 = getGL(); r.n1 = getGN();
        r.conv = getGL(); r.n2 = getGN(); r.lin2 = getGL();
        r.has_skip = skip; r.skip.W = nullptr; r.skip.b = nullptr;
        if (skip) r.skip = getGL();
        return r;
    };
    GLp gc0 = getGL();
    RBp gcrb[6]; gcrb[0] = getRB(1); for (int i = 1; i < 6; ++i) gcrb[i] = getRB(0);
    GLp sh_lin1 = getGL(), sh_lin2 = getGL(); GNp sh_gn = getGN(); GLp sh_lin3 = getGL();
    GLp gg0 = getGL();
    RBp ggrb[6]; ggrb[0] = getRB(1); for (int i = 1; i < 6; ++i) ggrb[i] = getRB(0);
    RBp grb1 = getRB(1), grb2 = getRB(1); GNp g_gn = getGN(); GLp g_lin = getGL();
    GNp cam_gn = getGN(); GLp cam_lin = getGL(); GLp cam_fc = getGL();
    const float* Abody = nx();
    const float* Agar  = nx();
    const float* refv  = nx();
    const float* grefv = nx();

    // ---- workspace bump allocator
    size_t off = 0;
    auto alloc = [&](size_t bytes) {
        void* p = (char*)d_ws + off;
        off = (off + bytes + 255) & ~(size_t)255;
        return p;
    };
    const size_t NG = N_GARM;
    __bf16* X0   = (__bf16*)alloc(2ull * BATCHN * NG * 1024);
    __bf16* OUTA = (__bf16*)alloc(2ull * BATCHN * NG * 512);
    __bf16* OUTB = (__bf16*)alloc(2ull * BATCHN * NG * 512);
    __bf16* T1   = (__bf16*)alloc(2ull * BATCHN * NG * 1024);
    __bf16* T2   = (__bf16*)alloc(2ull * BATCHN * NG * 256);
    __bf16* T3   = (__bf16*)alloc(2ull * BATCHN * NG * 256);
    __bf16* T4   = (__bf16*)alloc(2ull * BATCHN * NG * 256);
    __bf16* SKIP = (__bf16*)alloc(2ull * BATCHN * NG * 512);
    __bf16* H1   = (__bf16*)alloc(2ull * BATCHN * NG * 64);
    __bf16* H2   = (__bf16*)alloc(2ull * BATCHN * NG * 32);
    __bf16* CC   = (__bf16*)alloc(2ull * BATCHN * N_BODY * 32);
    float*  RPB  = (float*)alloc(4ull * NG * 1024);
    float*  IPB  = (float*)alloc(4ull * BATCHN * 1024);
    int*    AIDX = (int*)alloc(4ull * NG * 16);
    float*  AVAL = (float*)alloc(4ull * NG * 16);
    // remaining ws: bf16 weight arena (consumed by gemm() below)

    float* out_shape = (float*)d_out;                           // (16,3,1723)
    float* out_cam   = out_shape + (size_t)BATCHN * 3 * N_BODY; // (16,3)
    float* out_gar   = out_cam + BATCHN * 3;                    // (16,3,4000)

    auto cp32 = [](int c) { return (c + 31) & ~31; };

    // ---- helpers
    auto gemm = [&](const float* Wf, int M, int K, bool transpose,
                    const __bf16* Xp, int CpX, const float* bias,
                    const __bf16* resid, __bf16* Ybf, float* Ycf,
                    int Nv, int relu) {
        int Kp  = cp32(K);
        int CpY = cp32(M);
        __bf16* Wp = (__bf16*)alloc((size_t)M * Kp * sizeof(__bf16));
        int total = M * Kp;
        k_pack_w<<<(total + 255) / 256, 256, 0, stream>>>(Wf, Wp, M, K, Kp, transpose ? 1 : 0);
        dim3 grid((Nv + 63) / 64, (CpY + 127) / 128, BATCHN);
        k_gemm_wmma<<<grid, 128, 0, stream>>>(Wp, Xp, bias, resid, Ybf, Ycf,
                                              M, Kp, CpX, CpY, Nv, relu);
    };
    auto gnorm = [&](const __bf16* xp, __bf16* yp, GNp p, int C, int N, int relu) {
        int groups = C / 8;
        k_groupnorm<<<dim3(groups, BATCHN), 256, 0, stream>>>(
            xp, yp, p.g, p.be, C, cp32(C), N, groups, relu);
    };
    auto res_block = [&](const RBp& p, const __bf16* x, int Ci, int Co, int N,
                         __bf16* out) {
        int Ch = Co / 2;
        int Cpx = cp32(Ci), Cph = cp32(Ch), Cpo = cp32(Co);
        gnorm(x, T1, p.pre, Ci, N, 1);
        gemm(p.lin1.W, Ch, Ci, false, T1, Cpx, p.lin1.b, nullptr, T2, nullptr, N, 0);
        gnorm(T2, T3, p.n1, Ch, N, 1);
        gemm(p.conv.W, Ch, Ch, true, T3, Cph, nullptr, nullptr, T4, nullptr, N, 0);
        k_spmm<<<dim3((Ch + 127) / 128, N, BATCHN), 128, 0, stream>>>(
            T4, AIDX, AVAL, p.conv.b, T2, Ch, Cph, N);
        gnorm(T2, T3, p.n2, Ch, N, 1);
        const __bf16* resid = x;
        if (p.has_skip) {
            gemm(p.skip.W, Co, Ci, false, x, Cpx, p.skip.b, nullptr, SKIP, nullptr, N, 0);
            resid = SKIP;
        }
        gemm(p.lin2.W, Co, Ch, false, T3, Cph, p.lin2.b, resid, out, nullptr, N, 0);
    };
    auto run_branch = [&](GLp l0, RBp* rbs, const float* refvp, const float* Aptr,
                          int N) -> __bf16* {
        k_adj_compress<<<(N + 127) / 128, 128, 0, stream>>>(Aptr, N, AIDX, AVAL);
        k_refproj<<<(N * 1024 + 255) / 256, 256, 0, stream>>>(l0.W, l0.b, refvp, RPB, N);
        k_imgproj<<<dim3(8, BATCHN), 128, 0, stream>>>(l0.W, img, IPB);
        size_t total = (size_t)BATCHN * N * 1024;
        k_bcast_add<<<(unsigned)((total + 255) / 256), 256, 0, stream>>>(RPB, IPB, X0, N, total);
        const __bf16* x = X0; int Ci = 1024;
        __bf16* outs[2] = {OUTA, OUTB};
        for (int i = 0; i < 6; ++i) {
            __bf16* o = outs[i & 1];
            res_block(rbs[i], x, Ci, 512, N, o);
            x = o; Ci = 512;
        }
        return (__bf16*)x;   // == OUTB after 6 blocks
    };

    // ================= body branch =================
    __bf16* xb = run_branch(gc0, gcrb, refv, Abody, N_BODY);

    // shape head: relu(lin1) -> lin2 -> relu(gn) -> lin3 (f32 ch-first out)
    gemm(sh_lin1.W, 64, 512, false, xb, 512, sh_lin1.b, nullptr, H1, nullptr, N_BODY, 1);
    gemm(sh_lin2.W, 32, 64, false, H1, 64, sh_lin2.b, nullptr, H2, nullptr, N_BODY, 0);
    gnorm(H2, T4, sh_gn, 32, N_BODY, 1);
    gemm(sh_lin3.W, 3, 32, false, T4, 32, sh_lin3.b, nullptr, nullptr, out_shape, N_BODY, 0);

    // camera head: relu(gn(x)) -> relu(lin 1x512) -> fc (3 x N_BODY)
    gnorm(xb, T1, cam_gn, 512, N_BODY, 1);
    gemm(cam_lin.W, 1, 512, false, T1, 512, cam_lin.b, nullptr, CC, nullptr, N_BODY, 1);
    k_reduce_fc<<<dim3(3, BATCHN), 256, 0, stream>>>(CC, cam_fc.W, cam_fc.b, out_cam, N_BODY);

    // ================= garment branch =================
    __bf16* xg = run_branch(gg0, ggrb, grefv, Agar, N_GARM);   // in OUTB

    res_block(grb1, xg, 512, 64, N_GARM, OUTA);
    res_block(grb2, OUTA, 64, 32, N_GARM, OUTB);
    gnorm(OUTB, T4, g_gn, 32, N_GARM, 1);
    gemm(g_lin.W, 3, 32, false, T4, 32, g_lin.b, nullptr, nullptr, out_gar, N_GARM, 0);
}
